// TopKGroupedSharedMoE_34557306863837
// MI455X (gfx1250) — matrix operand order repaired and checked by reading, not compiled, not run on hardware
//
#include <hip/hip_runtime.h>
#include <math.h>

// ---- problem constants (match reference) ----
#define T_TOK 4096     // 2 * 2048 tokens
#define DM    1024     // d_model
#define HID   4096     // hidden
#define NE    8        // experts
#define TOPK  2
#define NG    4        // groups

typedef __attribute__((ext_vector_type(16))) __bf16 v16bf;
typedef __attribute__((ext_vector_type(2)))  __bf16 v2bf;
typedef __attribute__((ext_vector_type(2)))  float  v2f;
typedef __attribute__((ext_vector_type(8)))  float  v8f;

// ---- workspace layout (bytes) ----
#define OFF_CNT      ((size_t)0)                          // 8 ints
#define OFF_LIST_TS  ((size_t)256)                        // NE*T ints   (t*2+slot)
#define OFF_LIST_W   (OFF_LIST_TS + (size_t)NE*T_TOK*4)   // NE*T floats
#define OFF_PSCORE   (OFF_LIST_W  + (size_t)NE*T_TOK*4)   // T*NE floats
#define OFF_XB       (OFF_PSCORE  + (size_t)T_TOK*NE*4)   // T*DM bf16   (8 MB)
#define OFF_WFT      (OFF_XB      + (size_t)T_TOK*DM*2)   // HID*DM bf16 (8 MB, reused)
#define OFF_WGT      (OFF_WFT     + (size_t)HID*DM*2)     // HID*DM bf16 (8 MB, reused)
#define OFF_WPT      (OFF_WGT     + (size_t)HID*DM*2)     // DM*HID bf16 (8 MB, reused)
#define OFF_H        (OFF_WPT     + (size_t)DM*HID*2)     // T*HID bf16  (32 MB, reused)
#define OFF_STAGING  (OFF_H       + (size_t)T_TOK*HID*2)  // T*2*DM f32  (32 MB)
#define OFF_SHARED   (OFF_STAGING + (size_t)T_TOK*2*DM*4) // T*DM f32    (16 MB)

__device__ __forceinline__ unsigned short f2bf(float f) {
  return __builtin_bit_cast(unsigned short, (__bf16)f);   // hw cvt, RNE
}
__device__ __forceinline__ unsigned pack2(float lo, float hi) {
  v2f f = {lo, hi};
  v2bf b = __builtin_convertvector(f, v2bf);              // v_cvt pair
  return __builtin_bit_cast(unsigned, b);
}

// ============================ init ============================
__global__ void moe_init_kernel(int* cnt) {
  if (threadIdx.x < NE) cnt[threadIdx.x] = 0;
}

// =================== f32 -> bf16 convert (row-major) ===================
__launch_bounds__(256)
__global__ void moe_conv_kernel(const float* __restrict__ src,
                                unsigned short* __restrict__ dst) {
  size_t i = ((size_t)blockIdx.x * 256 + threadIdx.x) * 8;
  float4 a = *(const float4*)(src + i);
  float4 b = *(const float4*)(src + i + 4);
  uint4 o;
  o.x = pack2(a.x, a.y); o.y = pack2(a.z, a.w);
  o.z = pack2(b.x, b.y); o.w = pack2(b.z, b.w);
  *(uint4*)(dst + i) = o;
}

// =========== f32 [R][C] -> bf16 transposed [C][R] (LDS tiled) ===========
__launch_bounds__(256)
__global__ void moe_convT_kernel(const float* __restrict__ src,
                                 unsigned short* __restrict__ dst,
                                 int R, int C) {
  __shared__ float tile[64][65];
  const int rb = blockIdx.x * 64, cb = blockIdx.y * 64;
  const int tid = threadIdx.x;
  {
    const int r = tid >> 2, cq = (tid & 3) * 16;
    const float* p = src + (size_t)(rb + r) * C + cb + cq;
#pragma unroll
    for (int i = 0; i < 4; ++i) {
      float4 v = *(const float4*)(p + i * 4);
      tile[r][cq + i * 4 + 0] = v.x; tile[r][cq + i * 4 + 1] = v.y;
      tile[r][cq + i * 4 + 2] = v.z; tile[r][cq + i * 4 + 3] = v.w;
    }
  }
  __syncthreads();
  {
    const int c = tid >> 2, rq = (tid & 3) * 16;
    unsigned o[8];
#pragma unroll
    for (int i = 0; i < 8; ++i)
      o[i] = pack2(tile[rq + 2 * i][c], tile[rq + 2 * i + 1][c]);
    unsigned short* q = dst + (size_t)(cb + c) * R + rb + rq;
    uint4 u0 = {o[0], o[1], o[2], o[3]}, u1 = {o[4], o[5], o[6], o[7]};
    *(uint4*)(q) = u0;
    *(uint4*)(q + 8) = u1;
  }
}

// ============================ router ============================
__launch_bounds__(256)
__global__ void moe_router_kernel(const float* __restrict__ x,
                                  const float* __restrict__ wr,
                                  int* __restrict__ cnt,
                                  int* __restrict__ list_ts,
                                  float* __restrict__ list_w,
                                  float* __restrict__ pscore) {
  const int lane = threadIdx.x & 31;
  const int t = blockIdx.x * 8 + (threadIdx.x >> 5);
  if (t >= T_TOK) return;
  const float* xr = x + (size_t)t * DM;

  float acc[NE];
#pragma unroll
  for (int e = 0; e < NE; ++e) acc[e] = 0.f;
  for (int k = lane; k < DM; k += 32) {
    float xv = xr[k];
#pragma unroll
    for (int e = 0; e < NE; ++e) acc[e] = fmaf(xv, wr[e * DM + k], acc[e]);
  }
#pragma unroll
  for (int e = 0; e < NE; ++e)
#pragma unroll
    for (int off = 16; off > 0; off >>= 1) acc[e] += __shfl_xor(acc[e], off, 32);

  if (lane != 0) return;

  float mx = acc[0];
#pragma unroll
  for (int e = 1; e < NE; ++e) mx = fmaxf(mx, acc[e]);
  float sc[NE], s = 0.f;
#pragma unroll
  for (int e = 0; e < NE; ++e) { sc[e] = __expf(acc[e] - mx); s += sc[e]; }
  float inv = 1.f / s;
#pragma unroll
  for (int e = 0; e < NE; ++e) { sc[e] *= inv; pscore[t * NE + e] = sc[e]; }

  float gs[NG];
#pragma unroll
  for (int g = 0; g < NG; ++g) gs[g] = fmaxf(sc[2 * g], sc[2 * g + 1]);
  int g1 = 0; float b1 = gs[0];
#pragma unroll
  for (int g = 1; g < NG; ++g) if (gs[g] > b1) { b1 = gs[g]; g1 = g; }
  int g2 = -1; float b2 = -3.4e38f;
#pragma unroll
  for (int g = 0; g < NG; ++g) if (g != g1 && gs[g] > b2) { b2 = gs[g]; g2 = g; }

  float msc[NE];
#pragma unroll
  for (int e = 0; e < NE; ++e) {
    int g = e >> 1;
    msc[e] = (g == g1 || g == g2) ? sc[e] : -3.4e38f;
  }
  int e1 = 0; float w1 = msc[0];
#pragma unroll
  for (int e = 1; e < NE; ++e) if (msc[e] > w1) { w1 = msc[e]; e1 = e; }
  int e2 = -1; float w2 = -3.4e38f;
#pragma unroll
  for (int e = 0; e < NE; ++e) if (e != e1 && msc[e] > w2) { w2 = msc[e]; e2 = e; }

  int p1 = atomicAdd(&cnt[e1], 1);
  list_ts[e1 * T_TOK + p1] = t * 2 + 0;
  list_w [e1 * T_TOK + p1] = w1;
  int p2 = atomicAdd(&cnt[e2], 1);
  list_ts[e2 * T_TOK + p2] = t * 2 + 1;
  list_w [e2 * T_TOK + p2] = w2;
}

// ====================== GEMM1: h = (x@wf) * silu(x@wg) =====================
// All-bf16 inputs (xb row-major [T][DM], weights transposed [HID][DM]).
// Tile 64x128, BK=64 (2 WMMA sub-steps), double-buffered LDS, fragment-order
// LDS image: loads are uint4, stores are ds_store_b128, no conversion VALU.
__launch_bounds__(256)
__global__ void moe_gemm1_kernel(const unsigned short* __restrict__ xb,
                                 const unsigned short* __restrict__ wfT,
                                 const unsigned short* __restrict__ wgT,
                                 unsigned short* __restrict__ hbuf,
                                 const int* __restrict__ cntp,
                                 const int* __restrict__ rowmap) {
  const int cnt = cntp ? *cntp : T_TOK;
  const int mb = blockIdx.x;
  if (mb * 64 >= cnt) return;
  const int nb = blockIdx.y;

  __shared__ alignas(16) unsigned short sA [2][2][4][512];
  __shared__ alignas(16) unsigned short sBf[2][2][8][512];
  __shared__ alignas(16) unsigned short sBg[2][2][8][512];

  const int tid = threadIdx.x, lane = tid & 31, wave = tid >> 5;
  const int wm = wave & 3, wn = wave >> 2;

  // A loader: row ra, k chunk aq (16 elems)
  const int ra = tid >> 2, aq = (tid & 3) * 16;
  const int arow = mb * 64 + ra;
  int tok;
  if (rowmap) tok = (arow < cnt) ? (rowmap[arow] >> 1) : 0;
  else        tok = arow;
  const unsigned short* ap = xb + (size_t)tok * DM + aq;

  // B loader: row n (0..127), k half kh (0 or 32)
  const int bn = tid & 127, kh = (tid >> 7) * 32;
  const unsigned short* bfp = wfT + (size_t)(nb * 128 + bn) * DM + kh;
  const unsigned short* bgp = wgT + (size_t)(nb * 128 + bn) * DM + kh;
  const int bsub = kh >> 5, bfr = bn >> 4, bnl = bn & 15;

  v8f accf[4] = {}, accg[4] = {};
  uint4 aR[2], bfR[4], bgR[4];

  auto loadTiles = [&](int ks) {
    const int off = ks * 64;
    aR[0] = *(const uint4*)(ap + off);
    aR[1] = *(const uint4*)(ap + off + 8);
#pragma unroll
    for (int i = 0; i < 4; ++i) {
      bfR[i] = *(const uint4*)(bfp + off + i * 8);
      bgR[i] = *(const uint4*)(bgp + off + i * 8);
    }
  };
  auto storeTiles = [&](int buf) {
#pragma unroll
    for (int u = 0; u < 2; ++u) {
      int k0 = aq + u * 8;
      int sub = k0 >> 5, lh = (k0 >> 4) & 1, vb = (k0 & 8) >> 1;  // vb in {0,4}
      *(uint4*)&sA[buf][sub][ra >> 4][((ra & 15) + 16 * lh) * 16 + vb * 2] = aR[u];
    }
#pragma unroll
    for (int i = 0; i < 4; ++i) {
      int l = bnl + ((i >= 2) ? 16 : 0);
      int vb2 = (i & 1) * 8;
      *(uint4*)&sBf[buf][bsub][bfr][l * 16 + vb2] = bfR[i];
      *(uint4*)&sBg[buf][bsub][bfr][l * 16 + vb2] = bgR[i];
    }
  };
  auto mma = [&](int buf) {
#pragma unroll
    for (int sub = 0; sub < 2; ++sub) {
      v16bf a = *(const v16bf*)&sA[buf][sub][wm][lane * 16];
      v16bf fB[4], gB[4];
#pragma unroll
      for (int j = 0; j < 4; ++j) {
        fB[j] = *(const v16bf*)&sBf[buf][sub][wn * 4 + j][lane * 16];
        gB[j] = *(const v16bf*)&sBg[buf][sub][wn * 4 + j][lane * 16];
      }
#pragma unroll
      for (int j = 0; j < 4; ++j) {
        accf[j] = __builtin_amdgcn_wmma_f32_16x16x32_bf16(false, a, false, fB[j],
                                                          (short)0, accf[j], false, false);
        accg[j] = __builtin_amdgcn_wmma_f32_16x16x32_bf16(false, a, false, gB[j],
                                                          (short)0, accg[j], false, false);
      }
    }
  };

  const int NSTEP = DM / 64;
  loadTiles(0);
  storeTiles(0);
  for (int ks = 0; ks < NSTEP; ++ks) {
    __syncthreads();
    if (ks + 1 < NSTEP) loadTiles(ks + 1);
    mma(ks & 1);
    if (ks + 1 < NSTEP) storeTiles((ks + 1) & 1);
  }

  // epilogue: h = fc * silu(gate) -> bf16 row-major [r][HID]
  const int mrow = mb * 64 + wm * 16 + ((lane >> 4) << 3);
  const int ncol0 = nb * 128 + wn * 64 + (lane & 15);
#pragma unroll
  for (int j = 0; j < 4; ++j) {
    int n = ncol0 + j * 16;
#pragma unroll
    for (int r = 0; r < 8; ++r) {
      float g = accg[j][r];
      float sl = g / (1.f + __expf(-g));
      hbuf[(size_t)(mrow + r) * HID + n] = f2bf(accf[j][r] * sl);
    }
  }
}

// ====================== GEMM2: out = (h @ wp) * route_w ====================
__launch_bounds__(256)
__global__ void moe_gemm2_kernel(const unsigned short* __restrict__ hbuf,
                                 const unsigned short* __restrict__ wpT, // [DM][HID]
                                 float* __restrict__ outbuf,
                                 const int* __restrict__ cntp,
                                 const int* __restrict__ rowts,
                                 const float* __restrict__ roww) {
  const int cnt = cntp ? *cntp : T_TOK;
  const int mb = blockIdx.x;
  if (mb * 64 >= cnt) return;
  const int nb = blockIdx.y;

  __shared__ alignas(16) unsigned short sA[2][2][4][512];
  __shared__ alignas(16) unsigned short sB[2][2][8][512];

  const int tid = threadIdx.x, lane = tid & 31, wave = tid >> 5;
  const int wm = wave & 3, wn = wave >> 2;

  const int ra = tid >> 2, aq = (tid & 3) * 16;
  const unsigned short* ap = hbuf + (size_t)(mb * 64 + ra) * HID + aq;

  const int bn = tid & 127, kh = (tid >> 7) * 32;
  const unsigned short* bp = wpT + (size_t)(nb * 128 + bn) * HID + kh;
  const int bsub = kh >> 5, bfr = bn >> 4, bnl = bn & 15;

  v8f acc[4] = {};
  uint4 aR[2], bR[4];

  auto loadTiles = [&](int ks) {
    const int off = ks * 64;
    aR[0] = *(const uint4*)(ap + off);
    aR[1] = *(const uint4*)(ap + off + 8);
#pragma unroll
    for (int i = 0; i < 4; ++i) bR[i] = *(const uint4*)(bp + off + i * 8);
  };
  auto storeTiles = [&](int buf) {
#pragma unroll
    for (int u = 0; u < 2; ++u) {
      int k0 = aq + u * 8;
      int sub = k0 >> 5, lh = (k0 >> 4) & 1, vb = (k0 & 8) >> 1;
      *(uint4*)&sA[buf][sub][ra >> 4][((ra & 15) + 16 * lh) * 16 + vb * 2] = aR[u];
    }
#pragma unroll
    for (int i = 0; i < 4; ++i) {
      int l = bnl + ((i >= 2) ? 16 : 0);
      int vb2 = (i & 1) * 8;
      *(uint4*)&sB[buf][bsub][bfr][l * 16 + vb2] = bR[i];
    }
  };
  auto mma = [&](int buf) {
#pragma unroll
    for (int sub = 0; sub < 2; ++sub) {
      v16bf a = *(const v16bf*)&sA[buf][sub][wm][lane * 16];
      v16bf B[4];
#pragma unroll
      for (int j = 0; j < 4; ++j)
        B[j] = *(const v16bf*)&sB[buf][sub][wn * 4 + j][lane * 16];
#pragma unroll
      for (int j = 0; j < 4; ++j)
        acc[j] = __builtin_amdgcn_wmma_f32_16x16x32_bf16(false, a, false, B[j],
                                                         (short)0, acc[j], false, false);
    }
  };

  const int NSTEP = HID / 64;
  loadTiles(0);
  storeTiles(0);
  for (int ks = 0; ks < NSTEP; ++ks) {
    __syncthreads();
    if (ks + 1 < NSTEP) loadTiles(ks + 1);
    mma(ks & 1);
    if (ks + 1 < NSTEP) storeTiles((ks + 1) & 1);
  }

  const int mrow = mb * 64 + wm * 16 + ((lane >> 4) << 3);
  const int ncol0 = nb * 128 + wn * 64 + (lane & 15);
#pragma unroll
  for (int j = 0; j < 4; ++j) {
    int n = ncol0 + j * 16;
#pragma unroll
    for (int r = 0; r < 8; ++r) {
      int row = mrow + r;
      if (row < cnt) {
        int ts = rowts ? rowts[row] : row;
        float w = roww ? roww[row] : 1.f;
        outbuf[(size_t)ts * DM + n] = w * acc[j][r];
      }
    }
  }
}

// ============================ combine / finalize ============================
__global__ void moe_combine_kernel(const float* __restrict__ sharedo,
                                   const float* __restrict__ staging,
                                   float* __restrict__ y) {
  unsigned i = blockIdx.x * 256u + threadIdx.x;
  unsigned t = i >> 10, d = i & 1023u;
  y[i] = sharedo[i] + staging[(size_t)t * 2048 + d] + staging[(size_t)t * 2048 + 1024 + d];
}

__global__ void moe_finalize_kernel(const int* __restrict__ cnt,
                                    const float* __restrict__ ps,
                                    float* __restrict__ out) {
  __shared__ float pe[NE];
  int e = threadIdx.x;
  if (e < NE) {
    float s = 0.f;
    for (int t = 0; t < T_TOK; ++t) s += ps[t * NE + e];  // fixed order
    pe[e] = s / (float)T_TOK;
    out[(size_t)T_TOK * DM + 1 + e] = (float)cnt[e];
  }
  __syncthreads();
  if (e == 0) {
    float loss = 0.f;
    for (int i = 0; i < NE; ++i) {
      float fe = (float)cnt[i] / (float)(T_TOK * TOPK);
      loss += fe * pe[i];
    }
    out[(size_t)T_TOK * DM] = (float)NE * loss - 1.f;
  }
}

// ============================ launch ============================
extern "C" void kernel_launch(void* const* d_in, const int* in_sizes, int n_in,
                              void* d_out, int out_size, void* d_ws, size_t ws_size,
                              hipStream_t stream) {
  const float* x   = (const float*)d_in[0];
  const float* wr  = (const float*)d_in[1];
  const float* wfc = (const float*)d_in[2];
  const float* wgt = (const float*)d_in[3];
  const float* wpj = (const float*)d_in[4];
  const float* wsf = (const float*)d_in[5];
  const float* wsg = (const float*)d_in[6];
  const float* wsp = (const float*)d_in[7];
  float* out = (float*)d_out;

  char* ws = (char*)d_ws;
  int*            cnt     = (int*)(ws + OFF_CNT);
  int*            list_ts = (int*)(ws + OFF_LIST_TS);
  float*          list_w  = (float*)(ws + OFF_LIST_W);
  float*          pscore  = (float*)(ws + OFF_PSCORE);
  unsigned short* xb      = (unsigned short*)(ws + OFF_XB);
  unsigned short* wfT     = (unsigned short*)(ws + OFF_WFT);
  unsigned short* wgT     = (unsigned short*)(ws + OFF_WGT);
  unsigned short* wpT     = (unsigned short*)(ws + OFF_WPT);
  unsigned short* hbuf    = (unsigned short*)(ws + OFF_H);
  float*          staging = (float*)(ws + OFF_STAGING);
  float*          sharedo = (float*)(ws + OFF_SHARED);

  moe_init_kernel<<<1, 32, 0, stream>>>(cnt);
  moe_router_kernel<<<T_TOK / 8, 256, 0, stream>>>(x, wr, cnt, list_ts, list_w, pscore);
  moe_conv_kernel<<<(T_TOK * DM) / (256 * 8), 256, 0, stream>>>(x, xb);

  dim3 blk(256);
  dim3 g1(T_TOK / 64, HID / 128);
  dim3 g2(T_TOK / 64, DM / 128);
  dim3 gtFC(DM / 64, HID / 64);   // [DM][HID] -> [HID][DM]
  dim3 gtPJ(HID / 64, DM / 64);   // [HID][DM] -> [DM][HID]

  // shared expert
  moe_convT_kernel<<<gtFC, blk, 0, stream>>>(wsf, wfT, DM, HID);
  moe_convT_kernel<<<gtFC, blk, 0, stream>>>(wsg, wgT, DM, HID);
  moe_convT_kernel<<<gtPJ, blk, 0, stream>>>(wsp, wpT, HID, DM);
  moe_gemm1_kernel<<<g1, blk, 0, stream>>>(xb, wfT, wgT, hbuf, nullptr, nullptr);
  moe_gemm2_kernel<<<g2, blk, 0, stream>>>(hbuf, wpT, sharedo, nullptr, nullptr, nullptr);

  // routed experts (weight buffers + h reused; stream-ordered)
  for (int e = 0; e < NE; ++e) {
    moe_convT_kernel<<<gtFC, blk, 0, stream>>>(wfc + (size_t)e * DM * HID, wfT, DM, HID);
    moe_convT_kernel<<<gtFC, blk, 0, stream>>>(wgt + (size_t)e * DM * HID, wgT, DM, HID);
    moe_convT_kernel<<<gtPJ, blk, 0, stream>>>(wpj + (size_t)e * HID * DM, wpT, HID, DM);
    moe_gemm1_kernel<<<g1, blk, 0, stream>>>(
        xb, wfT, wgT, hbuf, cnt + e, list_ts + (size_t)e * T_TOK);
    moe_gemm2_kernel<<<g2, blk, 0, stream>>>(
        hbuf, wpT, staging, cnt + e, list_ts + (size_t)e * T_TOK,
        list_w + (size_t)e * T_TOK);
  }

  moe_combine_kernel<<<(T_TOK * DM) / 256, 256, 0, stream>>>(sharedo, staging, out);
  moe_finalize_kernel<<<1, 32, 0, stream>>>(cnt, pscore, out);
}
